// GraphCosineSim_3435973837516
// MI455X (gfx1250) — matrix operand order repaired and checked by reading, not compiled, not run on hardware
//
#include <hip/hip_runtime.h>
#include <math.h>

// ---------------------------------------------------------------------------
// CDNA5 (gfx1250) wave32 WMMA types
// ---------------------------------------------------------------------------
typedef __attribute__((ext_vector_type(16))) __bf16        v16bf;
typedef __attribute__((ext_vector_type(8)))  float         v8f;
typedef __attribute__((ext_vector_type(4)))  unsigned int  v4u;

union FragU { v16bf v; v4u q[2]; };

static __device__ __forceinline__ v4u zero4() {
  v4u z = {0u, 0u, 0u, 0u};
  return z;
}
static __device__ __forceinline__ v8f zero8() {
  v8f z = {0.f, 0.f, 0.f, 0.f, 0.f, 0.f, 0.f, 0.f};
  return z;
}
static __device__ __forceinline__ v8f wmma_bf16(v16bf a, v16bf b, v8f c) {
  // D = A(16x32 bf16) * B(32x16 bf16) + C(16x16 f32)
  return __builtin_amdgcn_wmma_f32_16x16x32_bf16(false, a, false, b,
                                                 (short)0, c, false, false);
}
static __device__ __forceinline__ void store_bf2(__bf16* p, float a, float b) {
  union { __bf16 h[2]; unsigned u; } t;
  t.h[0] = (__bf16)a; t.h[1] = (__bf16)b;
  *(unsigned*)p = t.u;
}

// Problem constants: B=8, C=64, H=W=256, E=64, G=256, G/2=128, D=9
// ---------------------------------------------------------------------------
// Kernel 0: weight prep.
//   B-fragment layout (16-bit B, 32x16): lane = N column (n = lane&15);
//   lanes 0-15 hold K=0..7,16..23 (elements 0..15), lanes 16-31 hold K=8..15,24..31.
//   bp1: [t(9)][kt(8)][nt(8)][lane(32)][e(16)]  (Wg1: 128x256x3x3)
//   bp2: [t(9)][kt(4)][lane(32)][e(16)]         (Wg2: 9x128x3x3, N padded to 16)
//   bpX/bpY: [kt(2)][nt(4)][lane(32)][e(16)]    (Wx/Wy: 64x64)
//   bnSB: folded BN scale/bias per output channel (128)
// ---------------------------------------------------------------------------
__global__ __launch_bounds__(256) void prep_kernel(
    const float* __restrict__ Wx,  const float* __restrict__ Wy,
    const float* __restrict__ Wg1,
    const float* __restrict__ gma, const float* __restrict__ bta,
    const float* __restrict__ mean, const float* __restrict__ var,
    const float* __restrict__ Wg2,
    __bf16* __restrict__ bp1, __bf16* __restrict__ bp2,
    __bf16* __restrict__ bpX, __bf16* __restrict__ bpY,
    float2* __restrict__ bnSB)
{
  const int i = blockIdx.x * 256 + threadIdx.x;
  const int N1 = 9 * 8 * 8 * 32 * 16;   // 294912
  const int N2 = 9 * 4 * 32 * 16;       // 18432
  const int N3 = 2 * 2 * 4 * 32 * 16;   // 8192
  if (i < N1) {
    int e = i & 15, l = (i >> 4) & 31, nt = (i >> 9) & 7, kt = (i >> 12) & 7, t = i >> 15;
    int n = nt * 16 + (l & 15);
    int k = kt * 32 + ((e >> 3) << 4) + ((l >> 4) << 3) + (e & 7);
    bp1[i] = (__bf16)Wg1[(n * 256 + k) * 9 + t];
  } else if (i < N1 + N2) {
    int j = i - N1;
    int e = j & 15, l = (j >> 4) & 31, kt = (j >> 9) & 3, t = j >> 11;
    int n = l & 15;
    int k = kt * 32 + ((e >> 3) << 4) + ((l >> 4) << 3) + (e & 7);
    bp2[j] = (n < 9) ? (__bf16)Wg2[(n * 128 + k) * 9 + t] : (__bf16)0.f;
  } else if (i < N1 + N2 + N3) {
    int j = i - N1 - N2;
    int e = j & 15, l = (j >> 4) & 31, nt = (j >> 9) & 3, kt = (j >> 11) & 1, mat = (j >> 12) & 1;
    int n = nt * 16 + (l & 15);
    int k = kt * 32 + ((e >> 3) << 4) + ((l >> 4) << 3) + (e & 7);
    float v = mat ? Wy[n * 64 + k] : Wx[n * 64 + k];
    if (mat) bpY[j & 4095] = (__bf16)v; else bpX[j & 4095] = (__bf16)v;
  } else if (i < N1 + N2 + N3 + 128) {
    int n = i - N1 - N2 - N3;
    float s = gma[n] * rsqrtf(var[n] + 1e-5f);
    bnSB[n] = make_float2(s, bta[n] - mean[n] * s);
  }
}

// ---------------------------------------------------------------------------
// Kernel 1: embedding (two 1x1 convs via WMMA) + l2-normalize + r0 + gi.
// WG = 128 pixels of one image row (4 waves). Stages NCHW f32 -> LDS bf16,
// builds A-fragments (16-bit A 16x32 layout: lane half selects K-base 0/8),
// 32 WMMAs per wave, then per-pixel epilogue writes gi (NHWC bf16, 256 ch).
// ---------------------------------------------------------------------------
#define ESTRIDE 72   // bf16 row stride: 144 B (16-aligned rows, ok LDS banks)
__global__ __launch_bounds__(128) void embed_kernel(
    const float* __restrict__ x, const float* __restrict__ y,
    const __bf16* __restrict__ bpX, const __bf16* __restrict__ bpY,
    __bf16* __restrict__ gi, float* __restrict__ r0)
{
  __shared__ __bf16 sX[128 * ESTRIDE];
  __shared__ __bf16 sY[128 * ESTRIDE];
  const int tid = threadIdx.x;
  const int lane = tid & 31;
  const int wv = tid >> 5;                 // 0..3, owns M-tiles {2wv, 2wv+1}
  const int blk = blockIdx.x;
  const int b = blk >> 9;
  const int h = (blk >> 1) & 255;
  const int w0 = (blk & 1) << 7;
  const int mrow = lane & 15;
  const int hi = lane >> 4;
  const int kb0 = hi << 3;                 // A-frag K-base within 32-block

  const size_t plane = 65536;
  const float* xb = x + (size_t)b * 64 * plane + (size_t)h * 256 + w0;
  const float* yb = y + (size_t)b * 64 * plane + (size_t)h * 256 + w0;
  #pragma unroll 4
  for (int c = 0; c < 64; ++c) {
    sX[tid * ESTRIDE + c] = (__bf16)xb[(size_t)c * plane + tid];
    sY[tid * ESTRIDE + c] = (__bf16)yb[(size_t)c * plane + tid];
  }
  __syncthreads();

  v8f accX[2][4], accY[2][4];
  #pragma unroll
  for (int a = 0; a < 2; ++a)
    #pragma unroll
    for (int n = 0; n < 4; ++n) { accX[a][n] = zero8(); accY[a][n] = zero8(); }

  #pragma unroll
  for (int kt = 0; kt < 2; ++kt) {
    FragU aX[2], aY[2];
    #pragma unroll
    for (int mt = 0; mt < 2; ++mt) {
      const int pix = (2 * wv + mt) * 16 + mrow;
      const __bf16* px = &sX[pix * ESTRIDE + kt * 32 + kb0];
      const __bf16* py = &sY[pix * ESTRIDE + kt * 32 + kb0];
      aX[mt].q[0] = *(const v4u*)px;          // K kb0..kb0+7
      aX[mt].q[1] = *(const v4u*)(px + 16);   // K kb0+16..kb0+23
      aY[mt].q[0] = *(const v4u*)py;
      aY[mt].q[1] = *(const v4u*)(py + 16);
    }
    #pragma unroll
    for (int nt = 0; nt < 4; ++nt) {
      const v16bf bX = *(const v16bf*)(bpX + (size_t)((kt * 4 + nt) * 32 + lane) * 16);
      const v16bf bY = *(const v16bf*)(bpY + (size_t)((kt * 4 + nt) * 32 + lane) * 16);
      #pragma unroll
      for (int mt = 0; mt < 2; ++mt) {
        accX[mt][nt] = wmma_bf16(aX[mt].v, bX, accX[mt][nt]);
        accY[mt][nt] = wmma_bf16(aY[mt].v, bY, accY[mt][nt]);
      }
    }
  }
  __syncthreads();

  // D layout: element j -> M = j (lanes<16) / 8+j (lanes>=16); N = lane&15.
  #pragma unroll
  for (int mt = 0; mt < 2; ++mt)
    #pragma unroll
    for (int nt = 0; nt < 4; ++nt)
      #pragma unroll
      for (int j = 0; j < 8; ++j) {
        const int m = hi ? (8 + j) : j;
        const int pix = (2 * wv + mt) * 16 + m;
        const int ch = nt * 16 + mrow;
        sX[pix * ESTRIDE + ch] = (__bf16)accX[mt][nt][j];
        sY[pix * ESTRIDE + ch] = (__bf16)accY[mt][nt][j];
      }
  __syncthreads();

  // Per-pixel: normalize, cosine, r0, gi = [xn | yn | |xn-yn| | xn*yn]
  float sx = 0.f, sy = 0.f, sxy = 0.f;
  for (int c = 0; c < 64; ++c) {
    const float a = (float)sX[tid * ESTRIDE + c];
    const float bb = (float)sY[tid * ESTRIDE + c];
    sx += a * a; sy += bb * bb; sxy += a * bb;
  }
  const float nx = 1.f / (sqrtf(sx) + 1e-6f);
  const float ny = 1.f / (sqrtf(sy) + 1e-6f);
  const size_t gp = ((size_t)b << 16) + ((size_t)h << 8) + (size_t)(w0 + tid);
  r0[gp] = (sxy * nx * ny + 1.f) * 0.5f;
  __bf16* grow = gi + gp * 256;
  for (int c = 0; c < 64; c += 2) {
    const float a0 = (float)sX[tid * ESTRIDE + c] * nx;
    const float a1 = (float)sX[tid * ESTRIDE + c + 1] * nx;
    const float b0 = (float)sY[tid * ESTRIDE + c] * ny;
    const float b1 = (float)sY[tid * ESTRIDE + c + 1] * ny;
    store_bf2(grow + c,        a0, a1);
    store_bf2(grow + 64 + c,   b0, b1);
    store_bf2(grow + 128 + c,  fabsf(a0 - b0), fabsf(a1 - b1));
    store_bf2(grow + 192 + c,  a0 * b0, a1 * b1);
  }
}

// ---------------------------------------------------------------------------
// Kernel 2: conv3x3 256->128 (implicit GEMM, 309 GFLOP) + BN + ReLU.
// WG = 128 pixels x 128 out-ch, 8 waves in a 4(M) x 2(N) grid,
// 8 f32 accumulators (2 M-tiles x 4 N-tiles) per wave.
// K loop: 9 taps x 8 K-tiles of 32; A-fragments straight from gi (NHWC bf16),
// B-fragments from prepped weights (L2-resident).
// ---------------------------------------------------------------------------
__global__ __launch_bounds__(256) void conv1_kernel(
    const __bf16* __restrict__ gi, const __bf16* __restrict__ bp1,
    const float2* __restrict__ bnSB, __bf16* __restrict__ hOut)
{
  __shared__ __bf16 sH[128 * 136];   // out tile [pix][128ch + pad], 34 KB
  const int tid = threadIdx.x, lane = tid & 31, wv = tid >> 5;
  const int wm = wv & 3, wn = wv >> 2;
  const int blk = blockIdx.x;
  const int b = blk >> 9, h = (blk >> 1) & 255, w0 = (blk & 1) << 7;
  const int mrow = lane & 15, hi = lane >> 4, kb0 = hi << 3;

  v8f acc[2][4];
  #pragma unroll
  for (int a = 0; a < 2; ++a)
    #pragma unroll
    for (int n = 0; n < 4; ++n) acc[a][n] = zero8();

  for (int t = 0; t < 9; ++t) {
    const int dy = t / 3 - 1, dx = t % 3 - 1;
    const int hh = h + dy;
    if ((unsigned)hh >= 256u) continue;
    const __bf16* girow = gi + ((size_t)(b * 256 + hh) << 8) * 256;
    const int pw0 = w0 + (2 * wm) * 16 + mrow + dx;
    for (int kt = 0; kt < 8; ++kt) {
      FragU af[2];
      #pragma unroll
      for (int mt = 0; mt < 2; ++mt) {
        const int pw = pw0 + mt * 16;
        if ((unsigned)pw < 256u) {
          const __bf16* pa = girow + (size_t)pw * 256 + kt * 32 + kb0;
          af[mt].q[0] = *(const v4u*)pa;
          af[mt].q[1] = *(const v4u*)(pa + 16);
        } else {
          af[mt].q[0] = zero4();
          af[mt].q[1] = zero4();
        }
      }
      const __bf16* bb = bp1 + (size_t)(((t * 8 + kt) * 8 + 4 * wn) * 32) * 16;
      #pragma unroll
      for (int ntb = 0; ntb < 4; ++ntb) {
        const v16bf bf = *(const v16bf*)(bb + (size_t)(ntb * 32 + lane) * 16);
        acc[0][ntb] = wmma_bf16(af[0].v, bf, acc[0][ntb]);
        acc[1][ntb] = wmma_bf16(af[1].v, bf, acc[1][ntb]);
      }
    }
  }

  // BN + ReLU, repack through LDS for coalesced NHWC bf16 stores.
  #pragma unroll
  for (int ntb = 0; ntb < 4; ++ntb) {
    const int ch = (4 * wn + ntb) * 16 + mrow;
    const float2 sb = bnSB[ch];
    #pragma unroll
    for (int mt = 0; mt < 2; ++mt)
      #pragma unroll
      for (int j = 0; j < 8; ++j) {
        const int m = hi ? (8 + j) : j;
        const int pix = (2 * wm + mt) * 16 + m;
        float v = acc[mt][ntb][j] * sb.x + sb.y;
        sH[pix * 136 + ch] = (__bf16)(v > 0.f ? v : 0.f);
      }
  }
  __syncthreads();
  {
    const int pix = tid >> 1, half = tid & 1;
    const v4u* src = (const v4u*)&sH[pix * 136 + half * 64];
    const size_t gp = ((size_t)(b * 256 + h) << 8) + (size_t)(w0 + pix);
    v4u* dst = (v4u*)(hOut + gp * 128 + half * 64);
    #pragma unroll
    for (int q = 0; q < 8; ++q) dst[q] = src[q];
  }
}

// ---------------------------------------------------------------------------
// Kernel 3: conv3x3 128->9 + sigmoid + deg. N padded to 16 (cols 9..15 zero).
// WG = 128 pixels, 8 waves, 1 M-tile + 1 accumulator each.
// ---------------------------------------------------------------------------
__global__ __launch_bounds__(256) void conv2_kernel(
    const __bf16* __restrict__ hIn, const __bf16* __restrict__ bp2,
    const float* __restrict__ bg2, float* __restrict__ gates,
    float* __restrict__ deg)
{
  __shared__ float sL[128 * 17];
  const int tid = threadIdx.x, lane = tid & 31, wv = tid >> 5;
  const int blk = blockIdx.x;
  const int b = blk >> 9, h = (blk >> 1) & 255, w0 = (blk & 1) << 7;
  const int mrow = lane & 15, hi = lane >> 4, kb0 = hi << 3;

  v8f acc = zero8();
  for (int t = 0; t < 9; ++t) {
    const int dy = t / 3 - 1, dx = t % 3 - 1;
    const int hh = h + dy;
    if ((unsigned)hh >= 256u) continue;
    const __bf16* hrow = hIn + ((size_t)(b * 256 + hh) << 8) * 128;
    const int pw = w0 + wv * 16 + mrow + dx;
    const bool ok = (unsigned)pw < 256u;
    #pragma unroll
    for (int kt = 0; kt < 4; ++kt) {
      FragU af;
      if (ok) {
        const __bf16* pa = hrow + (size_t)pw * 128 + kt * 32 + kb0;
        af.q[0] = *(const v4u*)pa;
        af.q[1] = *(const v4u*)(pa + 16);
      } else {
        af.q[0] = zero4();
        af.q[1] = zero4();
      }
      const v16bf bf = *(const v16bf*)(bp2 + (size_t)((t * 4 + kt) * 32 + lane) * 16);
      acc = wmma_bf16(af.v, bf, acc);
    }
  }
  #pragma unroll
  for (int j = 0; j < 8; ++j) {
    const int m = hi ? (8 + j) : j;
    sL[(wv * 16 + m) * 17 + mrow] = acc[j];
  }
  __syncthreads();
  if (tid < 128) {
    const size_t gp = ((size_t)(b * 256 + h) << 8) + (size_t)(w0 + tid);
    float ds = 0.f, g[9];
    #pragma unroll
    for (int d = 0; d < 9; ++d) {
      const float v = sL[tid * 17 + d] + bg2[d];
      const float s = 1.f / (1.f + __expf(-v));
      g[d] = s; ds += s;
    }
    float* gr = gates + gp * 9;
    #pragma unroll
    for (int d = 0; d < 9; ++d) gr[d] = g[d];
    deg[gp] = ds;
  }
}

// ---------------------------------------------------------------------------
// Kernel 4: one gated propagation iteration (SHIFTS order). Final pass clips.
// ---------------------------------------------------------------------------
__global__ __launch_bounds__(256) void prop_kernel(
    const float* __restrict__ rIn, const float* __restrict__ gates,
    const float* __restrict__ deg, float* __restrict__ rOut, int doClip)
{
  const int gp = blockIdx.x * 256 + threadIdx.x;
  const int b = gp >> 16, h = (gp >> 8) & 255, w = gp & 255;
  const float* rb = rIn + ((size_t)b << 16);
  const float r = rb[(h << 8) + w];
  const float* gv = gates + (size_t)gp * 9;
  float agg = 0.f;
  constexpr int dyA[9] = {-1, -1, 0, 1, 1,  1,  0, -1, 0};
  constexpr int dxA[9] = { 0,  1, 1, 1, 0, -1, -1, -1, 0};
  #pragma unroll
  for (int d = 0; d < 9; ++d) {
    const int nh = h - dyA[d], nw = w - dxA[d];
    const float nv = ((unsigned)nh < 256u && (unsigned)nw < 256u)
                         ? rb[(nh << 8) + nw] : 0.f;
    agg += gv[d] * nv;
  }
  float rn = 0.3f * r + 0.7f * agg / (deg[gp] + 1e-6f);
  if (doClip) rn = fminf(fmaxf(rn, 0.f), 1.f);
  rOut[gp] = rn;
}

// ---------------------------------------------------------------------------
// Host launcher (graph-capture safe: launches only).
// Workspace (~408 MB, aliased):
//   [0, 256M)      gi (bf16)  -> after conv1, reused for gates (f32) + deg
//   [256M, 384M)   h  (bf16)
//   then rA, rB (f32), prepped weights, BN scale/bias.
// ---------------------------------------------------------------------------
extern "C" void kernel_launch(void* const* d_in, const int* in_sizes, int n_in,
                              void* d_out, int out_size, void* d_ws, size_t ws_size,
                              hipStream_t stream)
{
  (void)in_sizes; (void)n_in; (void)out_size; (void)ws_size;
  const float* x    = (const float*)d_in[0];
  const float* y    = (const float*)d_in[1];
  const float* Wx   = (const float*)d_in[2];
  const float* Wy   = (const float*)d_in[3];
  const float* Wg1  = (const float*)d_in[4];
  const float* gma  = (const float*)d_in[5];
  const float* bta  = (const float*)d_in[6];
  const float* mean = (const float*)d_in[7];
  const float* var  = (const float*)d_in[8];
  const float* Wg2  = (const float*)d_in[9];
  const float* bg2  = (const float*)d_in[10];

  char* ws = (char*)d_ws;
  const size_t SZ_GI = (size_t)8 * 65536 * 256 * 2;  // 268435456
  const size_t SZ_H  = (size_t)8 * 65536 * 128 * 2;  // 134217728
  const size_t SZ_R  = (size_t)8 * 65536 * 4;        // 2097152

  __bf16* gi   = (__bf16*)(ws);
  __bf16* hbuf = (__bf16*)(ws + SZ_GI);
  float*  rA   = (float*)(ws + SZ_GI + SZ_H);
  float*  rB   = (float*)(ws + SZ_GI + SZ_H + SZ_R);
  char*   tail = ws + SZ_GI + SZ_H + 2 * SZ_R;
  __bf16* bp1  = (__bf16*)(tail);                    // 589824 B
  __bf16* bp2  = (__bf16*)(tail + 589824);           // 36864 B
  __bf16* bpX  = (__bf16*)(tail + 589824 + 36864);   // 8192 B
  __bf16* bpY  = (__bf16*)(tail + 589824 + 36864 + 8192);
  float2* bnSB = (float2*)(tail + 589824 + 36864 + 2 * 8192);

  float* gates = (float*)(ws);                        // alias gi (dead after conv1)
  float* degA  = (float*)(ws + (size_t)8 * 65536 * 9 * 4);

  prep_kernel <<<1257, 256, 0, stream>>>(Wx, Wy, Wg1, gma, bta, mean, var, Wg2,
                                         bp1, bp2, bpX, bpY, bnSB);
  embed_kernel<<<4096, 128, 0, stream>>>(x, y, bpX, bpY, gi, rA);
  conv1_kernel<<<4096, 256, 0, stream>>>(gi, bp1, bnSB, hbuf);
  conv2_kernel<<<4096, 256, 0, stream>>>(hbuf, bp2, bg2, gates, degA);
  prop_kernel <<<2048, 256, 0, stream>>>(rA, gates, degA, rB, 0);
  prop_kernel <<<2048, 256, 0, stream>>>(rB, gates, degA, (float*)d_out, 1);
}
